// nmODE_61237643706979
// MI455X (gfx1250) — compile-verified, matrix operand order checked by reading, not run on hardware
//
#include <hip/hip_runtime.h>
#include <stdint.h>

// ---------- problem constants ----------
#define ROWS      25600      // B*N
#define ROWS_PAD  25616      // padded to 16 for per-batch 7x16 row tiles
#define DHID      256
#define DFF       1024
#define DMLP      128

// ---------- workspace layout (bytes, 256-aligned) ----------
#define OFF_FEATBF  0u                         // 25616*256*2  = 13,115,392
#define OFF_YTBF    13115392u                  // 25600*256*2  = 13,107,200
#define OFF_X2MAX   26222592u                  // 256*1024*4   =  1,048,576
#define OFF_S       27271168u                  // 256*128*4    =    131,072
#define OFF_H3      27402240u                  // 25600*128*4  = 13,107,200
#define OFF_W2BF    40509440u                  // 1024*256*2   =    524,288
#define OFF_W3BBF   41033728u                  // 128*256*2    =     65,536
// total ~41.1 MB

typedef __attribute__((ext_vector_type(16))) __bf16 v16bf;
typedef __attribute__((ext_vector_type(8)))  float  v8f;

union Frag { v16bf v; uint4 u[2]; };

__device__ __forceinline__ unsigned short f2bf(float f) {
  union { float f; unsigned u; } x; x.f = f;
  unsigned u = x.u;
  u += 0x7FFFu + ((u >> 16) & 1u);          // round-to-nearest-even
  return (unsigned short)(u >> 16);
}

// ---------- K1: feature = relu(x @ W1^T + b1)  -> bf16 (padded rows = 0) ----------
__global__ void feat_kernel(const float* __restrict__ x, const float* __restrict__ w1W,
                            const float* __restrict__ w1b,
                            unsigned short* __restrict__ featBF) {
  int r = blockIdx.x, j = threadIdx.x;
  float v = 0.f;
  if (r < ROWS) {
    float x0 = x[r*3+0], x1 = x[r*3+1], x2 = x[r*3+2];
    v = fmaf(x0, w1W[j*3+0], fmaf(x1, w1W[j*3+1], fmaf(x2, w1W[j*3+2], w1b[j])));
    v = fmaxf(v, 0.f);
  }
  featBF[r*DHID + j] = f2bf(v);
}

// ---------- K2: convert W2 (all) and yt-part of W3 to bf16 ----------
__global__ void convw_kernel(const float* __restrict__ w2W, const float* __restrict__ w3W,
                             unsigned short* __restrict__ w2bf,
                             unsigned short* __restrict__ w3bbf) {
  int t = blockIdx.x*256 + threadIdx.x;
  if (t < DFF*DHID) {
    w2bf[t] = f2bf(w2W[t]);
  } else {
    int s = t - DFF*DHID;                  // s < 128*256
    int j = s >> 8, k = s & 255;
    w3bbf[s] = f2bf(w3W[j*1280 + 1024 + k]);
  }
}

// ---------- K3: fused RK4 integration + linear time-interp + softmax pooling ----------
__global__ void rk4_kernel(const float* __restrict__ x, const float* __restrict__ w1W,
                           const float* __restrict__ w1b, const float* __restrict__ wpool,
                           unsigned short* __restrict__ ytBF) {
  __shared__ float cc[21];
  int r = blockIdx.x, j = threadIdx.x;
  if (j == 0) {
    float w[10], mx = -1e30f;
    for (int t = 0; t < 10; ++t) { w[t] = wpool[t]; mx = fmaxf(mx, w[t]); }
    float s = 0.f;
    for (int t = 0; t < 10; ++t) { w[t] = __expf(w[t]-mx); s += w[t]; }
    float inv = 1.f / s;
    for (int g = 0; g < 21; ++g) cc[g] = 0.f;
    for (int t = 0; t < 10; ++t) {
      float pos = (float)t * (20.0f/9.0f);
      int idx = (int)floorf(pos); if (idx > 19) idx = 19;
      float frac = pos - (float)idx; if (frac > 1.f) frac = 1.f;
      float ww = w[t]*inv;
      cc[idx]   += ww*(1.f-frac);
      cc[idx+1] += ww*frac;
    }
  }
  __syncthreads();
  // recompute gamma in-register (dot-3 is free vs 80 sins)
  float x0 = x[r*3], x1 = x[r*3+1], x2 = x[r*3+2];
  float g = fmaf(x0, w1W[j*3], fmaf(x1, w1W[j*3+1], fmaf(x2, w1W[j*3+2], w1b[j])));
  g = fmaxf(g, 0.f);
  const float h = 0.1f;
  float y = 0.f, pool = 0.f;
  for (int s = 1; s <= 20; ++s) {
    float s1 = __sinf(y + g);              float k1 = s1*s1 - y;
    float y2 = fmaf(0.5f*h, k1, y);
    float s2 = __sinf(y2 + g);             float k2 = s2*s2 - y2;
    float y3 = fmaf(0.5f*h, k2, y);
    float s3 = __sinf(y3 + g);             float k3 = s3*s3 - y3;
    float y4 = fmaf(h, k3, y);
    float s4 = __sinf(y4 + g);             float k4 = s4*s4 - y4;
    y = fmaf(h/6.f, k1 + 2.f*(k2 + k3) + k4, y);
    pool = fmaf(cc[s], y, pool);
  }
  ytBF[r*DHID + j] = f2bf(pool);
}

// ---------- K4: x2max[b,j] = max_n relu(feature @ W2^T + b2)  (fused GEMM + max) ----------
// wave per (batch, 16-col tile); 7 row tiles of 16 (M=100 padded to 112, masked)
__global__ void gemm2max_kernel(const unsigned short* __restrict__ featBF,
                                const unsigned short* __restrict__ w2bf,
                                const float* __restrict__ w2b,
                                float* __restrict__ x2max) {
  const int lane = threadIdx.x & 31;
  const int wave = threadIdx.x >> 5;
  const int ct   = blockIdx.x * 4 + wave;    // 0..63 column tiles
  const int b    = blockIdx.y;               // 0..255
  const int jbase = ct * 16;
  const int ln   = lane & 15;
  const int khi  = (lane >> 4) ? 8 : 0;      // ISA 16-bit A/B lane-half K offset

  Frag bfrag[8];
  #pragma unroll
  for (int ks = 0; ks < 8; ++ks) {           // K = 256 = 8 x 32
    const unsigned short* p = w2bf + (jbase + ln)*DHID + ks*32 + khi;
    bfrag[ks].u[0] = *(const uint4*)(p);
    bfrag[ks].u[1] = *(const uint4*)(p + 16);
  }
  float b2v = w2b[jbase + ln];
  float m = 0.0f;                            // relu output >= 0 -> 0 is max identity
  for (int rt = 0; rt < 7; ++rt) {
    const int rowbase = b * 100 + rt * 16;
    v8f c = {};
    #pragma unroll
    for (int ks = 0; ks < 8; ++ks) {
      Frag a;
      const unsigned short* p = featBF + (rowbase + ln)*DHID + ks*32 + khi;
      a.u[0] = *(const uint4*)(p);
      a.u[1] = *(const uint4*)(p + 16);
      c = __builtin_amdgcn_wmma_f32_16x16x32_bf16(false, a.v, false, bfrag[ks].v,
                                                  (short)0, c, false, false);
    }
    #pragma unroll
    for (int rr = 0; rr < 8; ++rr) {
      int rib = rt*16 + rr + ((lane >> 4) << 3);   // row within batch
      float v = fmaxf(c[rr] + b2v, 0.0f);
      if (rib < 100) m = fmaxf(m, v);
    }
  }
  m = fmaxf(m, __shfl_xor(m, 16, 32));       // combine lane halves (same column)
  if (lane < 16) x2max[b*DFF + jbase + lane] = m;
}

// ---------- K5: S[b,j] = b3[j] + x2max[b,:] . W3a[j,:]   (per-batch 1024->128) ----------
__global__ void sgemm_small(const float* __restrict__ x2max, const float* __restrict__ w3W,
                            const float* __restrict__ w3b, float* __restrict__ S) {
  int t = blockIdx.x*256 + threadIdx.x;      // 32768
  int b = t >> 7, j = t & 127;
  const float4* xr = (const float4*)(x2max + b*DFF);
  const float4* wr = (const float4*)(w3W + j*1280);
  float acc = w3b[j];
  #pragma unroll 4
  for (int k = 0; k < 256; ++k) {
    float4 a = xr[k], w = wr[k];
    acc = fmaf(a.x, w.x, fmaf(a.y, w.y, fmaf(a.z, w.z, fmaf(a.w, w.w, acc))));
  }
  S[t] = acc;
}

// ---------- K6: h3 = relu(S[b] + yt @ W3b^T)   (bf16 WMMA) ----------
__global__ void gemm3_kernel(const unsigned short* __restrict__ ytBF,
                             const unsigned short* __restrict__ w3bbf,
                             const float* __restrict__ S, float* __restrict__ h3) {
  const int lane = threadIdx.x & 31;
  const int wave = threadIdx.x >> 5;
  const int ct   = blockIdx.y * 4 + wave;    // 0..7
  const int rowbase = blockIdx.x * 16;       // 1600 row tiles
  const int jbase   = ct * 16;
  const int ln  = lane & 15;
  const int khi = (lane >> 4) ? 8 : 0;

  v8f c = {};
  #pragma unroll
  for (int ks = 0; ks < 8; ++ks) {
    Frag a, bb;
    const unsigned short* pa = ytBF  + (rowbase + ln)*DHID + ks*32 + khi;
    a.u[0]  = *(const uint4*)(pa);  a.u[1]  = *(const uint4*)(pa + 16);
    const unsigned short* pb = w3bbf + (jbase + ln)*DHID + ks*32 + khi;
    bb.u[0] = *(const uint4*)(pb);  bb.u[1] = *(const uint4*)(pb + 16);
    c = __builtin_amdgcn_wmma_f32_16x16x32_bf16(false, a.v, false, bb.v,
                                                (short)0, c, false, false);
  }
  const int col = jbase + ln;
  #pragma unroll
  for (int rr = 0; rr < 8; ++rr) {
    int row = rowbase + rr + ((lane >> 4) << 3);
    int bb  = row / 100;
    h3[row*DMLP + col] = fmaxf(c[rr] + S[bb*DMLP + col], 0.0f);
  }
}

// ---------- K7: out = h3 @ out_W^T + out_b  (128 -> 2) ----------
__global__ void out_kernel(const float* __restrict__ h3, const float* __restrict__ outW,
                           const float* __restrict__ outb, float* __restrict__ out) {
  __shared__ float sw[256];
  __shared__ float sb[2];
  sw[threadIdx.x] = outW[threadIdx.x];
  if (threadIdx.x < 2) sb[threadIdx.x] = outb[threadIdx.x];
  __syncthreads();
  int row = blockIdx.x*256 + threadIdx.x;
  const float* hr = h3 + row*DMLP;
  float a0 = sb[0], a1 = sb[1];
  #pragma unroll 8
  for (int j = 0; j < 128; ++j) {
    float h = hr[j];
    a0 = fmaf(h, sw[j],       a0);
    a1 = fmaf(h, sw[128 + j], a1);
  }
  out[row*2]     = a0;
  out[row*2 + 1] = a1;
}

extern "C" void kernel_launch(void* const* d_in, const int* in_sizes, int n_in,
                              void* d_out, int out_size, void* d_ws, size_t ws_size,
                              hipStream_t stream) {
  (void)in_sizes; (void)n_in; (void)out_size; (void)ws_size;
  const float* x     = (const float*)d_in[0];
  const float* w1W   = (const float*)d_in[1];
  const float* w1b   = (const float*)d_in[2];
  const float* w2W   = (const float*)d_in[3];
  const float* w2b   = (const float*)d_in[4];
  const float* w3W   = (const float*)d_in[5];
  const float* w3b   = (const float*)d_in[6];
  const float* outW  = (const float*)d_in[7];
  const float* outb  = (const float*)d_in[8];
  const float* wpool = (const float*)d_in[9];

  char* ws = (char*)d_ws;
  unsigned short* featBF = (unsigned short*)(ws + OFF_FEATBF);
  unsigned short* ytBF   = (unsigned short*)(ws + OFF_YTBF);
  float*          x2max  = (float*)        (ws + OFF_X2MAX);
  float*          S      = (float*)        (ws + OFF_S);
  float*          h3     = (float*)        (ws + OFF_H3);
  unsigned short* w2bf   = (unsigned short*)(ws + OFF_W2BF);
  unsigned short* w3bbf  = (unsigned short*)(ws + OFF_W3BBF);

  float* out = (float*)d_out;

  feat_kernel<<<dim3(ROWS_PAD), dim3(256), 0, stream>>>(x, w1W, w1b, featBF);
  convw_kernel<<<dim3((DFF*DHID + DMLP*DHID)/256), dim3(256), 0, stream>>>(w2W, w3W, w2bf, w3bbf);
  rk4_kernel<<<dim3(ROWS), dim3(256), 0, stream>>>(x, w1W, w1b, wpool, ytBF);
  gemm2max_kernel<<<dim3(16, 256), dim3(128), 0, stream>>>(featBF, w2bf, w2b, x2max);
  sgemm_small<<<dim3(128), dim3(256), 0, stream>>>(x2max, w3W, w3b, S);
  gemm3_kernel<<<dim3(1600, 2), dim3(128), 0, stream>>>(ytBF, w3bbf, S, h3);
  out_kernel<<<dim3(100), dim3(256), 0, stream>>>(h3, outW, outb, out);
}
